// HiResPrecipNet_fl_2_32_smaller_57174604645057
// MI455X (gfx1250) — matrix operand order repaired and checked by reading, not compiled, not run on hardware
//
#include <hip/hip_runtime.h>
#include <hip/hip_bf16.h>

#define N_LOWC   60000
#define N_HIGHC  300000
#define E_L2HC   1600000
#define E_HHC    1600000
#define DFEAT    32
#define NEG_SLOPE 0.2f
#define BN_EPS   1e-5f

typedef __attribute__((ext_vector_type(16))) _Float16 v16h;
typedef __attribute__((ext_vector_type(8)))  _Float16 v8h;
typedef __attribute__((ext_vector_type(8)))  float    v8f;

union AFrag {
  v16h v;
  struct { v8h lo, hi; } p;
};

// ---------------------------------------------------------------- utilities
__global__ void fill_u32_kernel(unsigned* __restrict__ p, int n, unsigned v) {
  int i = blockIdx.x * blockDim.x + threadIdx.x;
  if (i < n) p[i] = v;
}

__device__ __forceinline__ unsigned enc_ord(float f) {
  unsigned b = __float_as_uint(f);
  return (b & 0x80000000u) ? ~b : (b | 0x80000000u);
}
__device__ __forceinline__ float dec_ord(unsigned u) {
  unsigned b = (u & 0x80000000u) ? (u & 0x7FFFFFFFu) : ~u;
  return __uint_as_float(b);
}

// ------------------------------------------------- WMMA dense node transform
// Y[N x FOUT] = X[N x K] @ W[K x FOUT] + bias   (W row-major as in reference)
// One wave -> 16-row tile. X tile staged to LDS as f16 (row stride 128),
// W staged column-major in LDS. Fragments assembled with ds_load_b128.
template <int FOUT>
__global__ __launch_bounds__(256) void wmma_linear_kernel(
    const float* __restrict__ X, int N, int K, int ldx,
    const float* __restrict__ W, const float* __restrict__ bias,
    float* __restrict__ Y, int ldy) {
  __shared__ _Float16 Wh[FOUT * 128];        // column-major: Wh[n*128 + k]
  __shared__ _Float16 Xh[8 * 16 * 128];      // per-wave 16-row tile, stride 128
  const int Kpad = (K + 31) & ~31;           // K <= 125 -> Kpad <= 128
  const int tid = threadIdx.x;

  // ---- stage W (zero-padded along K), column-major for contiguous-K reads
  for (int idx = tid; idx < FOUT * Kpad; idx += 256) {
    int n = idx / Kpad, k = idx - n * Kpad;
    Wh[n * 128 + k] = (k < K) ? (_Float16)W[k * FOUT + n] : (_Float16)0.0f;
  }

  const int lane = tid & 31;
  const int wave = tid >> 5;
  const int tile = blockIdx.x * 8 + wave;
  _Float16* xw = &Xh[wave * (16 * 128)];

  // ---- stage this wave's 16 x Kpad X tile as f16.
  // Clamped addresses -> unconditional loads; zeroing via selects (no branches).
  const int pairsPerRow = Kpad >> 1;
  for (int p = lane; p < 16 * pairsPerRow; p += 32) {
    int r = p / pairsPerRow;
    int k = (p - r * pairsPerRow) * 2;
    int row = tile * 16 + r;
    int crow = row < N ? row : N - 1;
    const float* base = X + (size_t)crow * ldx;
    int k0c = k < K ? k : K - 1;
    int k1c = (k + 1) < K ? (k + 1) : K - 1;
    float v0 = base[k0c];
    float v1 = base[k1c];
    bool rv = row < N;
    v0 = (rv && k < K) ? v0 : 0.0f;
    v1 = (rv && (k + 1) < K) ? v1 : 0.0f;
    union { _Float16 h[2]; unsigned u; } pk;
    pk.h[0] = (_Float16)v0;
    pk.h[1] = (_Float16)v1;
    *(unsigned*)&xw[r * 128 + k] = pk.u;
  }
  __syncthreads();

  const int m = lane & 15;
  const int hi = (lane >> 4) & 1;
  const int kbA = hi ? 16 : 0;   // A 16x32 f16 layout (contiguous K, matches B)
  const int kbB = hi ? 16 : 0;   // B 32x16 f16 layout
  const _Float16* arow = &xw[m * 128];
  const _Float16* bcol0 = &Wh[m * 128];
  const _Float16* bcol1 = &Wh[((FOUT == 32 ? 16 : 0) + m) * 128];

  v8f c0 = {};
  v8f c1 = {};
  for (int k0 = 0; k0 < Kpad; k0 += 32) {
    AFrag a, b0, b1;
    a.p.lo = *(const v8h*)(arow + k0 + kbA);        // K = k0+kbA .. +7
    a.p.hi = *(const v8h*)(arow + k0 + kbA + 8);    // K = k0+kbA+8 .. +15
    b0.p.lo = *(const v8h*)(bcol0 + k0 + kbB);      // K = k0+kbB .. +7
    b0.p.hi = *(const v8h*)(bcol0 + k0 + kbB + 8);  // K = k0+kbB+8 .. +15
    c0 = __builtin_amdgcn_wmma_f32_16x16x32_f16(false, a.v, false, b0.v,
                                                (short)0, c0, false, false);
    if (FOUT == 32) {
      b1.p.lo = *(const v8h*)(bcol1 + k0 + kbB);
      b1.p.hi = *(const v8h*)(bcol1 + k0 + kbB + 8);
      c1 = __builtin_amdgcn_wmma_f32_16x16x32_f16(false, a.v, false, b1.v,
                                                  (short)0, c1, false, false);
    }
  }

  const int mbase = hi ? 8 : 0;
#pragma unroll
  for (int r = 0; r < 8; ++r) {
    int orow = tile * 16 + mbase + r;
    if (orow < N) {
      Y[(size_t)orow * ldy + m] = c0[r] + bias[m];
      if (FOUT == 32) Y[(size_t)orow * ldy + 16 + m] = c1[r] + bias[16 + m];
    }
  }
}

// ------------------------------------------------------------- edge kernels
// Self-loops: edges e >= E use src=dst=e-E (reference appends arange(N)).
__global__ void edge_score_kernel(const float* __restrict__ hl,
                                  const float* __restrict__ hr,
                                  const int* __restrict__ src,
                                  const int* __restrict__ dst,
                                  const float* __restrict__ att,
                                  float* __restrict__ score,
                                  unsigned* __restrict__ mkey, int E, int nloop) {
  const int ET = E + nloop;
  const int stride = gridDim.x * blockDim.x;
  for (int e = blockIdx.x * blockDim.x + threadIdx.x; e < ET; e += stride) {
    int en = e + stride;
    if (en < E) {
      __builtin_prefetch(&src[en], 0, 0);
      __builtin_prefetch(&dst[en], 0, 0);
    }
    int s, d;
    if (e < E) { s = src[e]; d = dst[e]; } else { s = d = e - E; }
    const float* a = hl + (size_t)s * DFEAT;
    const float* b = hr + (size_t)d * DFEAT;
    float acc = 0.f;
#pragma unroll
    for (int k = 0; k < DFEAT; ++k) {
      float t = a[k] + b[k];
      t = (t > 0.f) ? t : NEG_SLOPE * t;
      acc += t * att[k];
    }
    score[e] = acc;
    atomicMax(&mkey[d], enc_ord(acc));
  }
}

__global__ void edge_exp_kernel(const int* __restrict__ dst,
                                const unsigned* __restrict__ mkey,
                                float* __restrict__ score,
                                float* __restrict__ den,
                                float* __restrict__ deg, int E, int nloop) {
  const int ET = E + nloop;
  const int stride = gridDim.x * blockDim.x;
  for (int e = blockIdx.x * blockDim.x + threadIdx.x; e < ET; e += stride) {
    int d = (e < E) ? dst[e] : e - E;
    float m = dec_ord(mkey[d]);
    float ex = __expf(score[e] - m);
    score[e] = ex;
    atomicAdd(&den[d], ex);
    atomicAdd(&deg[d], 1.0f);
  }
}

__global__ void edge_agg_kernel(const float* __restrict__ hl,
                                const int* __restrict__ src,
                                const int* __restrict__ dst,
                                const float* __restrict__ score,
                                const float* __restrict__ den,
                                float* __restrict__ agg, int E, int nloop) {
  const int ET = E + nloop;
  const int stride = gridDim.x * blockDim.x;
  for (int e = blockIdx.x * blockDim.x + threadIdx.x; e < ET; e += stride) {
    int en = e + stride;
    if (en < E) {
      __builtin_prefetch(&src[en], 0, 0);
      __builtin_prefetch(&score[en], 0, 0);
    }
    int s, d;
    if (e < E) { s = src[e]; d = dst[e]; } else { s = d = e - E; }
    float alpha = score[e] / den[d];
    const float* a = hl + (size_t)s * DFEAT;
    float* o = agg + (size_t)d * DFEAT;
#pragma unroll
    for (int k = 0; k < DFEAT; ++k) atomicAdd(&o[k], alpha * a[k]);
  }
}

__global__ void node_finalize_kernel(const float* __restrict__ agg,
                                     const float* __restrict__ deg,
                                     const float* __restrict__ bias,
                                     float* __restrict__ Y, int ldy, int coff,
                                     int N) {
  int idx = blockIdx.x * blockDim.x + threadIdx.x;
  if (idx < N * DFEAT) {
    int n = idx / DFEAT, k = idx % DFEAT;
    Y[(size_t)n * ldy + coff + k] = agg[idx] / fmaxf(deg[n], 1.f) + bias[k];
  }
}

__global__ void copy_col0_kernel(const float* __restrict__ z,
                                 float* __restrict__ Y, int ldy, int N) {
  int n = blockIdx.x * blockDim.x + threadIdx.x;
  if (n < N) Y[(size_t)n * ldy] = z[n];
}

// -------------------------------------------------------------- batch norm
__global__ void bn_stats_kernel(const float* __restrict__ X, int N, int F,
                                float* __restrict__ sums /* 2F */) {
  extern __shared__ float sh[];
  for (int i = threadIdx.x; i < 2 * F; i += blockDim.x) sh[i] = 0.f;
  __syncthreads();
  const int total = N * F;
  for (int idx = blockIdx.x * blockDim.x + threadIdx.x; idx < total;
       idx += gridDim.x * blockDim.x) {
    int f = idx % F;
    float v = X[idx];
    atomicAdd(&sh[f], v);
    atomicAdd(&sh[F + f], v * v);
  }
  __syncthreads();
  for (int i = threadIdx.x; i < 2 * F; i += blockDim.x) atomicAdd(&sums[i], sh[i]);
}

__global__ void bn_apply_kernel(const float* __restrict__ X,
                                float* __restrict__ Y, int N, int F,
                                const float* __restrict__ sums,
                                const float* __restrict__ g,
                                const float* __restrict__ b, int relu) {
  int idx = blockIdx.x * blockDim.x + threadIdx.x;
  if (idx < N * F) {
    int f = idx % F;
    float invN = 1.f / (float)N;
    float mu = sums[f] * invN;
    float var = sums[F + f] * invN - mu * mu;
    float sc = g[f] * rsqrtf(var + BN_EPS);
    float v = (X[idx] - mu) * sc + b[f];
    if (relu) v = fmaxf(v, 0.f);
    Y[idx] = v;
  }
}

// -------------------------------------------------------------- output head
__global__ void fc_head_kernel(const float* __restrict__ H /* N x 16, pre-relu */,
                               const float* __restrict__ W2,
                               const float* __restrict__ b2,
                               float* __restrict__ out, int N) {
  int n = blockIdx.x * blockDim.x + threadIdx.x;
  if (n < N) {
    const float* h = H + (size_t)n * 16;
    float s = b2[0];
#pragma unroll
    for (int k = 0; k < 16; ++k) s += fmaxf(h[k], 0.f) * W2[k];
    out[n] = s;
  }
}

// =========================================================================
extern "C" void kernel_launch(void* const* d_in, const int* in_sizes, int n_in,
                              void* d_out, int out_size, void* d_ws,
                              size_t ws_size, hipStream_t stream) {
  (void)in_sizes; (void)n_in; (void)out_size; (void)ws_size;
  const float* x_low   = (const float*)d_in[0];
  const float* x_high  = (const float*)d_in[1];
  const float* z_std   = (const float*)d_in[2];
  const float* l2h_Wl  = (const float*)d_in[3];
  const float* l2h_bl  = (const float*)d_in[4];
  const float* l2h_Wr  = (const float*)d_in[5];
  const float* l2h_br  = (const float*)d_in[6];
  const float* l2h_att = (const float*)d_in[7];
  const float* l2h_bias= (const float*)d_in[8];
  const float* bn1_g   = (const float*)d_in[9];
  const float* bn1_b   = (const float*)d_in[10];
  const float* gWl[3]  = {(const float*)d_in[11], (const float*)d_in[17], (const float*)d_in[23]};
  const float* gbl[3]  = {(const float*)d_in[12], (const float*)d_in[18], (const float*)d_in[24]};
  const float* gWr[3]  = {(const float*)d_in[13], (const float*)d_in[19], (const float*)d_in[25]};
  const float* gbr[3]  = {(const float*)d_in[14], (const float*)d_in[20], (const float*)d_in[26]};
  const float* gatt[3] = {(const float*)d_in[15], (const float*)d_in[21], (const float*)d_in[27]};
  const float* gbias[3]= {(const float*)d_in[16], (const float*)d_in[22], (const float*)d_in[28]};
  const float* bn_g[3] = {(const float*)d_in[29], (const float*)d_in[31], (const float*)d_in[33]};
  const float* bn_b[3] = {(const float*)d_in[30], (const float*)d_in[32], (const float*)d_in[34]};
  const float* fc1_W   = (const float*)d_in[35];
  const float* fc1_b   = (const float*)d_in[36];
  const float* fc2_W   = (const float*)d_in[37];
  const float* fc2_b   = (const float*)d_in[38];
  const int* l2h_src   = (const int*)d_in[39];
  const int* l2h_dst   = (const int*)d_in[40];
  const int* hh_src    = (const int*)d_in[41];
  const int* hh_dst    = (const int*)d_in[42];
  float* out = (float*)d_out;

  // ---- workspace carve-out (floats), 256B-aligned chunks
  float* ws = (float*)d_ws;
  size_t off = 0;
  auto alloc = [&](size_t nf) {
    float* p = ws + off;
    off += (nf + 63) & ~(size_t)63;
    return p;
  };
  float* hl_low = alloc((size_t)N_LOWC * DFEAT);
  float* hl     = alloc((size_t)N_HIGHC * DFEAT);   // also reused as fc1 out (N x 16)
  float* hr     = alloc((size_t)N_HIGHC * DFEAT);
  float* agg    = alloc((size_t)N_HIGHC * DFEAT);
  float* bufA   = alloc((size_t)N_HIGHC * 33);
  float* bufB   = alloc((size_t)N_HIGHC * 33);
  float* den    = alloc(N_HIGHC);
  float* deg    = alloc(N_HIGHC);
  unsigned* mkey = (unsigned*)alloc(N_HIGHC);
  float* score  = alloc((size_t)E_HHC + N_HIGHC);   // >= E_L2H too
  float* stats  = alloc(128);                        // 2*F sums

  const int TB = 256;
  auto blocks_for = [&](long n) { return (int)((n + TB - 1) / TB); };
  auto zero_u32 = [&](void* p, long n) {
    fill_u32_kernel<<<blocks_for(n), TB, 0, stream>>>((unsigned*)p, (int)n, 0u);
  };
  auto linear = [&](int FOUT, const float* X, int N, int K, int ldx,
                    const float* W, const float* b, float* Y, int ldy) {
    int tiles = (N + 15) / 16;
    int blks = (tiles + 7) / 8;  // 8 waves (tiles) per 256-thread block
    if (FOUT == 32)
      wmma_linear_kernel<32><<<blks, 256, 0, stream>>>(X, N, K, ldx, W, b, Y, ldy);
    else
      wmma_linear_kernel<16><<<blks, 256, 0, stream>>>(X, N, K, ldx, W, b, Y, ldy);
  };
  auto gat_edges = [&](const float* hls, const int* src, const int* dst,
                       const float* att, const float* bias, int E, int nloop,
                       float* Y, int ldy, int coff) {
    zero_u32(mkey, N_HIGHC);
    zero_u32(den, N_HIGHC);
    zero_u32(deg, N_HIGHC);
    zero_u32(agg, (long)N_HIGHC * DFEAT);
    long ET = (long)E + nloop;
    int eblks = blocks_for(ET);
    if (eblks > 16384) eblks = 16384;
    edge_score_kernel<<<eblks, TB, 0, stream>>>(hls, hr, src, dst, att, score,
                                                mkey, E, nloop);
    edge_exp_kernel<<<eblks, TB, 0, stream>>>(dst, mkey, score, den, deg, E, nloop);
    edge_agg_kernel<<<eblks, TB, 0, stream>>>(hls, src, dst, score, den, agg, E, nloop);
    node_finalize_kernel<<<blocks_for((long)N_HIGHC * DFEAT), TB, 0, stream>>>(
        agg, deg, bias, Y, ldy, coff, N_HIGHC);
  };
  auto batchnorm = [&](const float* X, float* Y, int F, const float* g,
                       const float* b, int relu) {
    zero_u32(stats, 2 * F);
    bn_stats_kernel<<<512, TB, 2 * F * sizeof(float), stream>>>(X, N_HIGHC, F, stats);
    bn_apply_kernel<<<blocks_for((long)N_HIGHC * F), TB, 0, stream>>>(
        X, Y, N_HIGHC, F, stats, g, b, relu);
  };

  // ---- low -> high bipartite GATv2 encoder
  linear(32, x_low, N_LOWC, 125, 125, l2h_Wl, l2h_bl, hl_low, DFEAT);
  linear(32, x_high, N_HIGHC, 1, 1, l2h_Wr, l2h_br, hr, DFEAT);
  gat_edges(hl_low, l2h_src, l2h_dst, l2h_att, l2h_bias, E_L2HC, /*nloop=*/0,
            bufA, /*ldy=*/33, /*coff=*/1);
  copy_col0_kernel<<<blocks_for(N_HIGHC), TB, 0, stream>>>(z_std, bufA, 33, N_HIGHC);

  // ---- BN1 (no relu), then 3 high-high GATv2 layers with BN+relu between
  batchnorm(bufA, bufB, 33, bn1_g, bn1_b, /*relu=*/0);

  int Kin[3] = {33, 32, 32};
  const float* xin = bufB;
  for (int i = 0; i < 3; ++i) {
    linear(32, xin, N_HIGHC, Kin[i], Kin[i], gWl[i], gbl[i], hl, DFEAT);
    linear(32, xin, N_HIGHC, Kin[i], Kin[i], gWr[i], gbr[i], hr, DFEAT);
    gat_edges(hl, hh_src, hh_dst, gatt[i], gbias[i], E_HHC, /*nloop=*/N_HIGHC,
              bufA, /*ldy=*/32, /*coff=*/0);
    batchnorm(bufA, bufB, 32, bn_g[i], bn_b[i], /*relu=*/1);
    xin = bufB;
  }

  // ---- head: fc1 (WMMA, 32->16) -> relu -> fc2 (16->1)
  linear(16, bufB, N_HIGHC, 32, 32, fc1_W, fc1_b, hl, 16);
  fc_head_kernel<<<blocks_for(N_HIGHC), TB, 0, stream>>>(hl, fc2_W, fc2_b, out,
                                                         N_HIGHC);
}
